// DecoderBlock_28587302322808
// MI455X (gfx1250) — matrix operand order repaired and checked by reading, not compiled, not run on hardware
//
#include <hip/hip_runtime.h>
#include <hip/hip_bf16.h>

typedef __bf16 bh16;
typedef __attribute__((ext_vector_type(16))) __bf16 v16bf;
typedef __attribute__((ext_vector_type(8)))  __bf16 v8bf;
typedef __attribute__((ext_vector_type(8)))  float  v8f;

#define HEADS_C 8
#define NST     512
#define SEQ     2048
#define DEP     64
#define BATCH   2
#define SPAD    (SEQ + 32)
#define ROWS    (BATCH * SEQ)
#define FFN     (4 * NST)

// ---------------------------------------------------------------- helpers

__device__ __forceinline__ v8f zero8() {
  v8f z;
#pragma unroll
  for (int i = 0; i < 8; ++i) z[i] = 0.0f;
  return z;
}

__device__ __forceinline__ v8f wmma_bf16f32(v16bf a, v16bf b, v8f c) {
  // (neg_a, A, neg_b, B, c_mod, C, reuse_a, reuse_b)
  return __builtin_amdgcn_wmma_f32_16x16x32_bf16(false, a, false, b, (short)0, c,
                                                 false, false);
}

// A fragment: 16x32 bf16 tile, row-major source with leading dim `ld`.
// lane L: row = L&15, half = L>>4; K window = [half*16 .. +15] contiguous
__device__ __forceinline__ v16bf load_a_frag(const bh16* base, int ld) {
  const int lane = threadIdx.x & 31;
  const int row = lane & 15, hv = lane >> 4;
  const bh16* p = base + row * ld + hv * 16;
  v8bf lo = *(const v8bf*)p;
  v8bf hi = *(const v8bf*)(p + 8);
  v16bf r;
#pragma unroll
  for (int i = 0; i < 8; ++i) { r[i] = lo[i]; r[i + 8] = hi[i]; }
  return r;
}

// B fragment from Bt (N-major rows over K): 16 rows (n) x 32 cols (k).
// lane L: n = L&15, K window = [half*16 .. +15] contiguous.
__device__ __forceinline__ v16bf load_b_frag(const bh16* base, int ld) {
  const int lane = threadIdx.x & 31;
  const int nrow = lane & 15, hv = lane >> 4;
  const bh16* p = base + nrow * ld + hv * 16;
  v8bf lo = *(const v8bf*)p;
  v8bf hi = *(const v8bf*)(p + 8);
  v16bf r;
#pragma unroll
  for (int i = 0; i < 8; ++i) { r[i] = lo[i]; r[i + 8] = hi[i]; }
  return r;
}

// B fragment with per-lane row clamp (for E-window edge tiles)
__device__ __forceinline__ v16bf load_b_rows_clamped(const bh16* mat, int ld,
                                                     int row0, int maxrow, int koff) {
  const int lane = threadIdx.x & 31;
  int rr = row0 + (lane & 15);
  if (rr > maxrow) rr = maxrow;
  const bh16* p = mat + rr * ld + koff + (lane >> 4) * 16;
  v8bf lo = *(const v8bf*)p;
  v8bf hi = *(const v8bf*)(p + 8);
  v16bf r;
#pragma unroll
  for (int i = 0; i < 8; ++i) { r[i] = lo[i]; r[i + 8] = hi[i]; }
  return r;
}

// one wave computes a 16(M) x 64(N) strip over K, software-pipelined:
// next k-step fragments are issued before the current step's WMMAs so the
// backend emits partial LOADcnt waits and overlaps VMEM with matrix ops.
__device__ __forceinline__ void gemm_acc16x64(const bh16* __restrict__ A, int lda,
                                              const bh16* __restrict__ Bt, int ldbt,
                                              int K, v8f acc[4]) {
  v16bf a  = load_a_frag(A, lda);
  v16bf b0 = load_b_frag(Bt + (size_t)0  * ldbt, ldbt);
  v16bf b1 = load_b_frag(Bt + (size_t)16 * ldbt, ldbt);
  v16bf b2 = load_b_frag(Bt + (size_t)32 * ldbt, ldbt);
  v16bf b3 = load_b_frag(Bt + (size_t)48 * ldbt, ldbt);
  for (int k = 32; k < K; k += 32) {
    v16bf an  = load_a_frag(A + k, lda);
    v16bf bn0 = load_b_frag(Bt + (size_t)0  * ldbt + k, ldbt);
    v16bf bn1 = load_b_frag(Bt + (size_t)16 * ldbt + k, ldbt);
    v16bf bn2 = load_b_frag(Bt + (size_t)32 * ldbt + k, ldbt);
    v16bf bn3 = load_b_frag(Bt + (size_t)48 * ldbt + k, ldbt);
    acc[0] = wmma_bf16f32(a, b0, acc[0]);
    acc[1] = wmma_bf16f32(a, b1, acc[1]);
    acc[2] = wmma_bf16f32(a, b2, acc[2]);
    acc[3] = wmma_bf16f32(a, b3, acc[3]);
    a = an; b0 = bn0; b1 = bn1; b2 = bn2; b3 = bn3;
  }
  acc[0] = wmma_bf16f32(a, b0, acc[0]);
  acc[1] = wmma_bf16f32(a, b1, acc[1]);
  acc[2] = wmma_bf16f32(a, b2, acc[2]);
  acc[3] = wmma_bf16f32(a, b3, acc[3]);
}

// ---------------------------------------------------------------- prep kernels

// Coalesced fp32 -> bf16 transpose: WT[n*K + k] = W[k*N + n], 32x32 LDS tiles.
__global__ void __launch_bounds__(256)
transpose_bf16_kernel(const float* __restrict__ W, bh16* __restrict__ WT,
                      int K, int N) {
  __shared__ bh16 tile[32][33];
  const int kx = blockIdx.x * 32;       // K base
  const int nx = blockIdx.y * 32;       // N base
  const int tx = threadIdx.x & 31, ty = threadIdx.x >> 5;   // ty: 0..7
#pragma unroll
  for (int i = 0; i < 4; ++i) {
    int kk = kx + ty + 8 * i;
    tile[ty + 8 * i][tx] = (bh16)W[(size_t)kk * N + nx + tx];
  }
  __syncthreads();
#pragma unroll
  for (int i = 0; i < 4; ++i) {
    int nn = nx + ty + 8 * i;
    WT[(size_t)nn * K + kx + tx] = tile[tx][ty + 8 * i];
  }
}

__global__ void convert_bf16_kernel(const float* __restrict__ src, bh16* __restrict__ dst,
                                    int n) {
  int idx = blockIdx.x * 256 + threadIdx.x;
  if (idx < n) dst[idx] = (bh16)src[idx];
}

// ---------------------------------------------------------------- layernorm

__global__ void __launch_bounds__(128)
ln_kernel(const float* __restrict__ X, const float* __restrict__ gam,
          const float* __restrict__ bet, bh16* __restrict__ out) {
  const int row = blockIdx.x;
  const int tid = threadIdx.x;
  const float* x = X + (size_t)row * NST;
  float v0 = x[tid], v1 = x[tid + 128], v2 = x[tid + 256], v3 = x[tid + 384];
  float s = v0 + v1 + v2 + v3;
#pragma unroll
  for (int off = 1; off < 32; off <<= 1) s += __shfl_xor(s, off, 32);
  __shared__ float r1[4], r2[4];
  int wid = tid >> 5, lane = tid & 31;
  if (lane == 0) r1[wid] = s;
  __syncthreads();
  float mu = (r1[0] + r1[1] + r1[2] + r1[3]) * (1.0f / NST);
  float d0 = v0 - mu, d1 = v1 - mu, d2 = v2 - mu, d3 = v3 - mu;
  float q = d0 * d0 + d1 * d1 + d2 * d2 + d3 * d3;
#pragma unroll
  for (int off = 1; off < 32; off <<= 1) q += __shfl_xor(q, off, 32);
  if (lane == 0) r2[wid] = q;
  __syncthreads();
  float rstd = rsqrtf((r2[0] + r2[1] + r2[2] + r2[3]) * (1.0f / NST) + 1e-5f);
  bh16* o = out + (size_t)row * NST;
  o[tid]       = (bh16)(d0 * rstd * gam[tid]       + bet[tid]);
  o[tid + 128] = (bh16)(d1 * rstd * gam[tid + 128] + bet[tid + 128]);
  o[tid + 256] = (bh16)(d2 * rstd * gam[tid + 256] + bet[tid + 256]);
  o[tid + 384] = (bh16)(d3 * rstd * gam[tid + 384] + bet[tid + 384]);
}

// ---------------------------------------------------------------- GEMM kernels

__global__ void __launch_bounds__(256)
gemm_qkv_kernel(const bh16* __restrict__ A, const bh16* __restrict__ Bt,
                const float* __restrict__ bias,
                bh16* __restrict__ qo, bh16* __restrict__ ko, bh16* __restrict__ vT) {
  int w = threadIdx.x >> 5;
  int m0 = blockIdx.x * 128 + w * 16;
  int n0 = blockIdx.y * 64;
  v8f acc[4] = {zero8(), zero8(), zero8(), zero8()};
  gemm_acc16x64(A + (size_t)m0 * NST, NST, Bt + (size_t)n0 * NST, NST, NST, acc);
  int lane = threadIdx.x & 31;
  int nl = lane & 15, mb = (lane >> 4) * 8;
#pragma unroll
  for (int j = 0; j < 4; ++j)
#pragma unroll
    for (int r = 0; r < 8; ++r) {
      int m = m0 + mb + r, n = n0 + j * 16 + nl;
      float vv = acc[j][r] + bias[n];
      int b = m >> 11, s = m & (SEQ - 1);
      int d = n & 63;
      if (n < 512) {
        int h = n >> 6;
        qo[(((size_t)(b * HEADS_C + h)) * SEQ + s) * DEP + d] = (bh16)vv;
      } else if (n < 1024) {
        int h = (n - 512) >> 6;
        ko[(((size_t)(b * HEADS_C + h)) * SEQ + s) * DEP + d] = (bh16)vv;
      } else {
        int h = (n - 1024) >> 6;
        vT[(((size_t)(b * HEADS_C + h)) * DEP + d) * SPAD + s] = (bh16)vv;
      }
    }
}

__global__ void __launch_bounds__(256)
gemm_proj_kernel(const bh16* __restrict__ A, const bh16* __restrict__ Bt,
                 const float* __restrict__ bias, const float* __restrict__ xres,
                 float* __restrict__ x1) {
  int w = threadIdx.x >> 5;
  int m0 = blockIdx.x * 128 + w * 16;
  int n0 = blockIdx.y * 64;
  v8f acc[4] = {zero8(), zero8(), zero8(), zero8()};
  gemm_acc16x64(A + (size_t)m0 * NST, NST, Bt + (size_t)n0 * NST, NST, NST, acc);
  int lane = threadIdx.x & 31;
  int nl = lane & 15, mb = (lane >> 4) * 8;
#pragma unroll
  for (int j = 0; j < 4; ++j)
#pragma unroll
    for (int r = 0; r < 8; ++r) {
      int m = m0 + mb + r, n = n0 + j * 16 + nl;
      size_t idx = (size_t)m * NST + n;
      x1[idx] = xres[idx] + acc[j][r] + bias[n];
    }
}

__global__ void __launch_bounds__(256)
gemm_ffn1_kernel(const bh16* __restrict__ A, const bh16* __restrict__ Bt,
                 const float* __restrict__ bias, bh16* __restrict__ hout) {
  int w = threadIdx.x >> 5;
  int m0 = blockIdx.x * 128 + w * 16;
  int n0 = blockIdx.y * 64;
  v8f acc[4] = {zero8(), zero8(), zero8(), zero8()};
  gemm_acc16x64(A + (size_t)m0 * NST, NST, Bt + (size_t)n0 * NST, NST, NST, acc);
  int lane = threadIdx.x & 31;
  int nl = lane & 15, mb = (lane >> 4) * 8;
#pragma unroll
  for (int j = 0; j < 4; ++j)
#pragma unroll
    for (int r = 0; r < 8; ++r) {
      int m = m0 + mb + r, n = n0 + j * 16 + nl;
      float v = acc[j][r] + bias[n];
      // tanh-approx GELU (jax.nn.gelu default)
      float g = 0.5f * v * (1.0f + tanhf(0.7978845608f * (v + 0.044715f * v * v * v)));
      hout[(size_t)m * FFN + n] = (bh16)g;
    }
}

__global__ void __launch_bounds__(256)
gemm_ffn2_kernel(const bh16* __restrict__ A, const bh16* __restrict__ Bt,
                 const float* __restrict__ bias, const float* __restrict__ x1,
                 float* __restrict__ out) {
  int w = threadIdx.x >> 5;
  int m0 = blockIdx.x * 128 + w * 16;
  int n0 = blockIdx.y * 64;
  v8f acc[4] = {zero8(), zero8(), zero8(), zero8()};
  gemm_acc16x64(A + (size_t)m0 * FFN, FFN, Bt + (size_t)n0 * FFN, FFN, FFN, acc);
  int lane = threadIdx.x & 31;
  int nl = lane & 15, mb = (lane >> 4) * 8;
#pragma unroll
  for (int j = 0; j < 4; ++j)
#pragma unroll
    for (int r = 0; r < 8; ++r) {
      int m = m0 + mb + r, n = n0 + j * 16 + nl;
      size_t idx = (size_t)m * NST + n;
      out[idx] = x1[idx] + acc[j][r] + bias[n];
    }
}

// ---------------------------------------------------------------- flash attention
// grid: B*H*(SEQ/16)/4 blocks of 128 threads (4 waves); each wave owns a 16-row
// query block and streams 32-key chunks with online softmax. Causal mask is
// computed analytically; the skewed relative term rel[s,t] = q[s].E[t+S-1-s]
// is produced per-chunk via 3 extra WMMA tiles staged through LDS.
// k- and v-fragment loads are hoisted to the top of each chunk so their VMEM
// latency overlaps the score WMMAs and the softmax VALU work.
__global__ void __launch_bounds__(128)
attn_kernel(const bh16* __restrict__ q, const bh16* __restrict__ k,
            const bh16* __restrict__ vT, const bh16* __restrict__ E,
            bh16* __restrict__ att) {
  __shared__ float relS[4][16 * 48];
  __shared__ __align__(16) bh16 pS[4][16 * 32];

  const int w = threadIdx.x >> 5;
  const int lane = threadIdx.x & 31;
  const int g = blockIdx.x * 4 + w;
  const int sblk = g & 127;            // SEQ/16 = 128
  const int bh = g >> 7;               // 0..15 = b*8+h
  const int s0 = sblk * 16;

  const bh16* qh = q + ((size_t)bh * SEQ + s0) * DEP;
  const bh16* kh = k + (size_t)bh * SEQ * DEP;
  const bh16* vh = vT + (size_t)bh * DEP * SPAD;
  const bh16* Eh = E + (size_t)(bh & (HEADS_C - 1)) * SEQ * DEP;

  const v16bf aq0 = load_a_frag(qh, DEP);
  const v16bf aq1 = load_a_frag(qh + 32, DEP);

  v8f o0 = zero8(), o1 = zero8(), o2 = zero8(), o3 = zero8();
  float mrow[8], lrow[8];
#pragma unroll
  for (int r = 0; r < 8; ++r) { mrow[r] = -1e30f; lrow[r] = 0.0f; }

  const int nl = lane & 15, mb = (lane >> 4) * 8;

  for (int t0 = 0; t0 <= s0 + 15; t0 += 32) {
    // ---- issue all chunk-level global loads up front (k for scores, v for p.v)
    const v16bf kb00 = load_b_frag(kh + (size_t)t0 * DEP, DEP);
    const v16bf kb01 = load_b_frag(kh + (size_t)t0 * DEP + 32, DEP);
    const v16bf kb10 = load_b_frag(kh + (size_t)(t0 + 16) * DEP, DEP);
    const v16bf kb11 = load_b_frag(kh + (size_t)(t0 + 16) * DEP + 32, DEP);
    const v16bf vb0 = load_b_frag(vh + (size_t)0  * SPAD + t0, SPAD);
    const v16bf vb1 = load_b_frag(vh + (size_t)16 * SPAD + t0, SPAD);
    const v16bf vb2 = load_b_frag(vh + (size_t)32 * SPAD + t0, SPAD);
    const v16bf vb3 = load_b_frag(vh + (size_t)48 * SPAD + t0, SPAD);

    // ---- scores = q . k^T (16x32), K = DEP = 64 -> 4 WMMAs
    v8f sc0 = zero8(), sc1 = zero8();
    sc0 = wmma_bf16f32(aq0, kb00, sc0);
    sc0 = wmma_bf16f32(aq1, kb01, sc0);
    sc1 = wmma_bf16f32(aq0, kb10, sc1);
    sc1 = wmma_bf16f32(aq1, kb11, sc1);

    // ---- relative term window: j = t + (SEQ-1) - s  ->  jb .. jb+46
    const int jb = t0 + (SEQ - 16) - s0;   // >= 0 always
#pragma unroll
    for (int f = 0; f < 3; ++f) {
      v8f rw = zero8();
      rw = wmma_bf16f32(aq0, load_b_rows_clamped(Eh, DEP, jb + f * 16, SEQ - 1, 0), rw);
      rw = wmma_bf16f32(aq1, load_b_rows_clamped(Eh, DEP, jb + f * 16, SEQ - 1, 32), rw);
#pragma unroll
      for (int r = 0; r < 8; ++r)
        relS[w][(mb + r) * 48 + f * 16 + nl] = rw[r];
    }
    asm volatile("s_wait_dscnt 0" ::: "memory");

    // ---- assemble scores (+rel, scale, causal mask), online softmax
#pragma unroll
    for (int r = 0; r < 8; ++r) {
      const int m = mb + r, srow = s0 + m;
      float v0 = (sc0[r] + relS[w][m * 48 + (nl + 15 - m)]) * 0.125f;
      float v1 = (sc1[r] + relS[w][m * 48 + (nl + 31 - m)]) * 0.125f;
      if (t0 + nl > srow)      v0 = -1e30f;
      if (t0 + 16 + nl > srow) v1 = -1e30f;
      float mx = fmaxf(v0, v1);
#pragma unroll
      for (int off = 1; off < 16; off <<= 1) mx = fmaxf(mx, __shfl_xor(mx, off, 32));
      const float newm = fmaxf(mrow[r], mx);
      const float alpha = __expf(mrow[r] - newm);
      const float p0 = __expf(v0 - newm), p1 = __expf(v1 - newm);
      float ls = p0 + p1;
#pragma unroll
      for (int off = 1; off < 16; off <<= 1) ls += __shfl_xor(ls, off, 32);
      lrow[r] = lrow[r] * alpha + ls;
      mrow[r] = newm;
      o0[r] *= alpha; o1[r] *= alpha; o2[r] *= alpha; o3[r] *= alpha;
      pS[w][m * 32 + nl]      = (bh16)p0;
      pS[w][m * 32 + 16 + nl] = (bh16)p1;
    }
    asm volatile("s_wait_dscnt 0" ::: "memory");

    // ---- accumulate o += p . v   (A = p 16x32 from LDS, Bt = vT rows = depth)
    const v16bf ap = load_a_frag(&pS[w][0], 32);
    o0 = wmma_bf16f32(ap, vb0, o0);
    o1 = wmma_bf16f32(ap, vb1, o1);
    o2 = wmma_bf16f32(ap, vb2, o2);
    o3 = wmma_bf16f32(ap, vb3, o3);
  }

  // ---- epilogue: normalize and scatter into (B,S, h*64+d) bf16 activation
  const int b = bh >> 3, h = bh & 7;
#pragma unroll
  for (int r = 0; r < 8; ++r) {
    const int m = mb + r, s = s0 + m;
    const float inv = 1.0f / lrow[r];
    bh16* dst = att + ((size_t)(b * SEQ + s)) * NST + h * DEP;
    dst[0  + nl] = (bh16)(o0[r] * inv);
    dst[16 + nl] = (bh16)(o1[r] * inv);
    dst[32 + nl] = (bh16)(o2[r] * inv);
    dst[48 + nl] = (bh16)(o3[r] * inv);
  }
}

// ---------------------------------------------------------------- launch

extern "C" void kernel_launch(void* const* d_in, const int* in_sizes, int n_in,
                              void* d_out, int out_size, void* d_ws, size_t ws_size,
                              hipStream_t stream) {
  const float* x     = (const float*)d_in[0];
  /* d_in[1] = mask: unused, causal mask computed analytically */
  const float* Wqkv  = (const float*)d_in[2];
  const float* bqkv  = (const float*)d_in[3];
  const float* Wproj = (const float*)d_in[4];
  const float* bproj = (const float*)d_in[5];
  const float* E     = (const float*)d_in[6];
  const float* g1    = (const float*)d_in[7];
  const float* b1    = (const float*)d_in[8];
  const float* g2    = (const float*)d_in[9];
  const float* b2    = (const float*)d_in[10];
  const float* W1    = (const float*)d_in[11];
  const float* bm1   = (const float*)d_in[12];
  const float* W2    = (const float*)d_in[13];
  const float* bm2   = (const float*)d_in[14];
  float* out = (float*)d_out;

  char* p = (char*)d_ws;
  auto carve = [&](size_t bytes) -> char* {
    char* r = p;
    p += (bytes + 255) & ~(size_t)255;
    return r;
  };
  bh16* abf    = (bh16*)carve((size_t)ROWS * NST * 2);
  bh16* WqkvT  = (bh16*)carve((size_t)(3 * NST) * NST * 2);
  bh16* WprojT = (bh16*)carve((size_t)NST * NST * 2);
  bh16* W1T    = (bh16*)carve((size_t)FFN * NST * 2);
  bh16* W2T    = (bh16*)carve((size_t)NST * FFN * 2);
  bh16* Ebf    = (bh16*)carve((size_t)HEADS_C * SEQ * DEP * 2);
  bh16* qbf    = (bh16*)carve((size_t)BATCH * HEADS_C * SEQ * DEP * 2);
  bh16* kbf    = (bh16*)carve((size_t)BATCH * HEADS_C * SEQ * DEP * 2);
  bh16* vTbf   = (bh16*)carve((size_t)BATCH * HEADS_C * DEP * SPAD * 2);
  bh16* attbf  = (bh16*)carve((size_t)ROWS * NST * 2);
  float* x1    = (float*)carve((size_t)ROWS * NST * 4);
  bh16* mbf    = (bh16*)carve((size_t)ROWS * NST * 2);
  bh16* hbf    = (bh16*)carve((size_t)ROWS * FFN * 2);

  // weight prep (bf16, transposed so WMMA B-operand reads are row-contiguous)
  transpose_bf16_kernel<<<dim3(NST / 32, (3 * NST) / 32), 256, 0, stream>>>(
      Wqkv, WqkvT, NST, 3 * NST);
  transpose_bf16_kernel<<<dim3(NST / 32, NST / 32), 256, 0, stream>>>(
      Wproj, WprojT, NST, NST);
  transpose_bf16_kernel<<<dim3(NST / 32, FFN / 32), 256, 0, stream>>>(
      W1, W1T, NST, FFN);
  transpose_bf16_kernel<<<dim3(FFN / 32, NST / 32), 256, 0, stream>>>(
      W2, W2T, FFN, NST);
  {
    int n = HEADS_C * SEQ * DEP;
    convert_bf16_kernel<<<(n + 255) / 256, 256, 0, stream>>>(E, Ebf, n);
  }

  // a = LN1(x) -> bf16
  ln_kernel<<<ROWS, 128, 0, stream>>>(x, g1, b1, abf);

  // qkv = a @ Wqkv + b, split into q / k / v^T per head
  gemm_qkv_kernel<<<dim3(ROWS / 128, (3 * NST) / 64), 256, 0, stream>>>(
      abf, WqkvT, bqkv, qbf, kbf, vTbf);

  // flash attention with relative positions
  attn_kernel<<<(BATCH * HEADS_C * (SEQ / 16)) / 4, 128, 0, stream>>>(
      qbf, kbf, vTbf, Ebf, attbf);

  // x1 = x + att @ Wproj + b
  gemm_proj_kernel<<<dim3(ROWS / 128, NST / 64), 256, 0, stream>>>(
      attbf, WprojT, bproj, x, x1);

  // m = LN2(x1) -> bf16
  ln_kernel<<<ROWS, 128, 0, stream>>>(x1, g2, b2, mbf);

  // h = gelu(m @ W1 + b) -> bf16
  gemm_ffn1_kernel<<<dim3(ROWS / 128, FFN / 64), 256, 0, stream>>>(
      mbf, W1T, bm1, hbf);

  // out = x1 + h @ W2 + b
  gemm_ffn2_kernel<<<dim3(ROWS / 128, NST / 64), 256, 0, stream>>>(
      hbf, W2T, bm2, x1, out);
}
